// GINENet_64622077936096
// MI455X (gfx1250) — matrix operand order repaired and checked by reading, not compiled, not run on hardware
//
#include <hip/hip_runtime.h>

// ---------------- CDNA5 WMMA types ----------------
typedef __attribute__((ext_vector_type(16))) _Float16 v16h;
typedef __attribute__((ext_vector_type(8)))  float    v8f;

#define N_NODES 100000
#define N_EDGES 1000000
#define N_GRAPHS 512
#define HID 64
#define LPAD 8                 // half-element row padding (keeps 16B alignment: 72*2=144B)
#define LSTR (HID + LPAD)

union HVec { v16h h; float4 f[2]; };
union E4  { float2 f2; _Float16 h[4]; };

// Fused 2-layer MLP: out = (relu(A @ w0 + b0)) @ w1 + b1, with modes:
//  MODE_IN : 0 -> A = in[row, 0..K_IN)  (zero-padded to 32)
//            1 -> A = in[row] + 1.1 * in2[row]   (GINE combine, K=64)
//  MODE_OUT: 0 -> out_f32 = y
//            1 -> out_f32 = relu(in2 + y)   (residual)
//            2 -> out_f16 = (f16)y          (edge features)
template<int K_IN, int MODE_IN, int MODE_OUT>
__global__ __launch_bounds__(128)
void mlp2_wmma(const float* __restrict__ in,  const float* __restrict__ in2,
               const float* __restrict__ w0,  const float* __restrict__ b0,
               const float* __restrict__ w1,  const float* __restrict__ b1,
               float* __restrict__ out_f32,   _Float16* __restrict__ out_f16,
               int nrows)
{
    __shared__ _Float16 W0T[HID][LSTR];   // transposed weights, f16
    __shared__ _Float16 W1T[HID][LSTR];
    __shared__ _Float16 Ah[16][LSTR];     // A row-tile, f16
    __shared__ _Float16 Zh[16][LSTR];     // hidden activations, f16

    const int tid  = threadIdx.x;
    const int lane = tid & 31;
    const int wv   = tid >> 5;        // wave id 0..3 -> output N tile
    const int hrow = lane & 15;       // A row / C column within tile
    const int hi   = lane >> 4;       // half-wave select

    // one-time: stage both weight matrices transposed into LDS as f16
    for (int i = tid; i < HID * HID; i += 128) {
        const int k = i >> 6, n = i & 63;
        W0T[n][k] = (_Float16)((k < K_IN) ? w0[k * HID + n] : 0.0f);
        W1T[n][k] = (_Float16)w1[k * HID + n];
    }

    const int   n_col = wv * 16 + hrow;   // this lane's output column
    const float bias0 = b0[n_col];
    const float bias1 = b1[n_col];
    const int   row_blk = blockIdx.x * 64;

    for (int rt = 0; rt < 4; ++rt) {
        const int row0 = row_blk + rt * 16;
        __syncthreads();                           // prior iter's LDS reads done
        // ---- stage A tile (16 rows x 64 halves) ----
        for (int i = tid; i < 16 * HID; i += 128) {
            const int r = i >> 6, c = i & 63;
            const int row = row0 + r;
            float v = 0.0f;
            if (row < nrows) {
                if constexpr (MODE_IN == 1) {
                    v = in[row * HID + c] + 1.1f * in2[row * HID + c];
                } else {
                    v = (c < K_IN) ? in[row * K_IN + c] : 0.0f;
                }
            }
            Ah[r][c] = (_Float16)v;
        }
        __syncthreads();

        // ---- GEMM1: z = A @ w0 ----
        v8f acc = {0.f,0.f,0.f,0.f,0.f,0.f,0.f,0.f};
        constexpr int KT1 = (K_IN + 31) / 32;
        #pragma unroll
        for (int kt = 0; kt < KT1; ++kt) {
            const int kb = kt * 32;
            HVec a, b;
            a.f[0] = *reinterpret_cast<const float4*>(&Ah[hrow][kb + hi * 8]);
            a.f[1] = *reinterpret_cast<const float4*>(&Ah[hrow][kb + hi * 8 + 16]);
            b.f[0] = *reinterpret_cast<const float4*>(&W0T[n_col][kb + hi * 8]);
            b.f[1] = *reinterpret_cast<const float4*>(&W0T[n_col][kb + hi * 8 + 16]);
            acc = __builtin_amdgcn_wmma_f32_16x16x32_f16(
                      false, a.h, false, b.h, (short)0, acc, false, false);
        }
        // epilogue1: bias + relu -> Zh (f16)
        #pragma unroll
        for (int r = 0; r < 8; ++r) {
            const int m = r + hi * 8;
            float z = acc[r] + bias0;
            Zh[m][n_col] = (_Float16)(z > 0.f ? z : 0.f);
        }
        __syncthreads();

        // ---- GEMM2: y = z @ w1  (K = 64) ----
        v8f acc2 = {0.f,0.f,0.f,0.f,0.f,0.f,0.f,0.f};
        #pragma unroll
        for (int kt = 0; kt < 2; ++kt) {
            const int kb = kt * 32;
            HVec a, b;
            a.f[0] = *reinterpret_cast<const float4*>(&Zh[hrow][kb + hi * 8]);
            a.f[1] = *reinterpret_cast<const float4*>(&Zh[hrow][kb + hi * 8 + 16]);
            b.f[0] = *reinterpret_cast<const float4*>(&W1T[n_col][kb + hi * 8]);
            b.f[1] = *reinterpret_cast<const float4*>(&W1T[n_col][kb + hi * 8 + 16]);
            acc2 = __builtin_amdgcn_wmma_f32_16x16x32_f16(
                       false, a.h, false, b.h, (short)0, acc2, false, false);
        }
        // epilogue2
        #pragma unroll
        for (int r = 0; r < 8; ++r) {
            const int m   = r + hi * 8;
            const int row = row0 + m;
            if (row < nrows) {
                float y = acc2[r] + bias1;
                if constexpr (MODE_OUT == 1) {
                    y += in2[row * HID + n_col];
                    out_f32[row * HID + n_col] = (y > 0.f ? y : 0.f);
                } else if constexpr (MODE_OUT == 2) {
                    out_f16[row * HID + n_col] = (_Float16)y;
                } else {
                    out_f32[row * HID + n_col] = y;
                }
            }
        }
    }
}

// msg = relu(h[src] + e); agg[dst] += msg   (16 threads/edge, 4 floats/thread)
__global__ __launch_bounds__(256)
void gine_scatter(const float* __restrict__ h, const _Float16* __restrict__ e,
                  const int* __restrict__ ei, float* __restrict__ agg, int nE)
{
    const int t    = blockIdx.x * 256 + threadIdx.x;
    const int edge = t >> 4;
    const int ch   = (t & 15) * 4;
    if (edge >= nE) return;
    const int s = ei[edge];
    const int d = ei[nE + edge];
    const float4 hv = *reinterpret_cast<const float4*>(&h[s * HID + ch]);
    E4 ev; ev.f2 = *reinterpret_cast<const float2*>(&e[edge * HID + ch]);
    float m0 = hv.x + (float)ev.h[0];
    float m1 = hv.y + (float)ev.h[1];
    float m2 = hv.z + (float)ev.h[2];
    float m3 = hv.w + (float)ev.h[3];
    float* ap = &agg[d * HID + ch];
    unsafeAtomicAdd(ap + 0, m0 > 0.f ? m0 : 0.f);
    unsafeAtomicAdd(ap + 1, m1 > 0.f ? m1 : 0.f);
    unsafeAtomicAdd(ap + 2, m2 > 0.f ? m2 : 0.f);
    unsafeAtomicAdd(ap + 3, m3 > 0.f ? m3 : 0.f);
}

// global_add_pool: out[batch[n]] += hN[n]
__global__ __launch_bounds__(256)
void pool_kernel(const float* __restrict__ hN, const int* __restrict__ batch,
                 float* __restrict__ out, int nN)
{
    const int t    = blockIdx.x * 256 + threadIdx.x;
    const int node = t >> 4;
    const int ch   = (t & 15) * 4;
    if (node >= nN) return;
    const int g = batch[node];
    const float4 v = *reinterpret_cast<const float4*>(&hN[node * HID + ch]);
    float* op = &out[g * HID + ch];
    unsafeAtomicAdd(op + 0, v.x);
    unsafeAtomicAdd(op + 1, v.y);
    unsafeAtomicAdd(op + 2, v.z);
    unsafeAtomicAdd(op + 3, v.w);
}

extern "C" void kernel_launch(void* const* d_in, const int* in_sizes, int n_in,
                              void* d_out, int out_size, void* d_ws, size_t ws_size,
                              hipStream_t stream)
{
    const float* x     = (const float*)d_in[0];
    const int*   ei    = (const int*)  d_in[1];
    const float* ea    = (const float*)d_in[2];
    const int*   batch = (const int*)  d_in[3];
    const float* wn0 = (const float*)d_in[4],  *bn0 = (const float*)d_in[5];
    const float* wn1 = (const float*)d_in[6],  *bn1 = (const float*)d_in[7];
    const float* we0 = (const float*)d_in[8],  *be0 = (const float*)d_in[9];
    const float* we1 = (const float*)d_in[10], *be1 = (const float*)d_in[11];
    const float* cw0 = (const float*)d_in[12], *cb0 = (const float*)d_in[13];
    const float* cw1 = (const float*)d_in[14], *cb1 = (const float*)d_in[15];
    const float* lw0 = (const float*)d_in[16], *lb0 = (const float*)d_in[17];
    const float* lw1 = (const float*)d_in[18], *lb1 = (const float*)d_in[19];

    // workspace: h (25.6MB f32) | agg (25.6MB f32) | e (128MB f16)
    float*    h   = (float*)d_ws;
    float*    agg = h + (size_t)N_NODES * HID;
    _Float16* e   = (_Float16*)(agg + (size_t)N_NODES * HID);

    const int mlpBlkN = (N_NODES + 63) / 64;   // 1563
    const int mlpBlkE = (N_EDGES + 63) / 64;   // 15625
    const int scatBlk = (N_EDGES * 16) / 256;  // 62500
    const int poolBlk = (N_NODES * 16 + 255) / 256;

    // h = MLP2(x)         [N,32] -> [N,64]
    mlp2_wmma<32, 0, 0><<<mlpBlkN, 128, 0, stream>>>(
        x, nullptr, wn0, bn0, wn1, bn1, h, nullptr, N_NODES);
    // e = MLP2(edge_attr) [E,16] -> [E,64], stored f16
    mlp2_wmma<16, 0, 2><<<mlpBlkE, 128, 0, stream>>>(
        ea, nullptr, we0, be0, we1, be1, nullptr, e, N_EDGES);

    // 3 residual GINE conv layers (h updated in place)
    for (int l = 0; l < 3; ++l) {
        hipMemsetAsync(agg, 0, (size_t)N_NODES * HID * sizeof(float), stream);
        gine_scatter<<<scatBlk, 256, 0, stream>>>(h, e, ei, agg, N_EDGES);
        mlp2_wmma<64, 1, 1><<<mlpBlkN, 128, 0, stream>>>(
            agg, h, cw0 + l * HID * HID, cb0 + l * HID,
            cw1 + l * HID * HID, cb1 + l * HID, h, nullptr, N_NODES);
    }

    // final GINE conv -> agg (in place over agg is block-local-safe)
    hipMemsetAsync(agg, 0, (size_t)N_NODES * HID * sizeof(float), stream);
    gine_scatter<<<scatBlk, 256, 0, stream>>>(h, e, ei, agg, N_EDGES);
    mlp2_wmma<64, 1, 0><<<mlpBlkN, 128, 0, stream>>>(
        agg, h, lw0, lb0, lw1, lb1, agg, nullptr, N_NODES);

    // global add pool
    hipMemsetAsync(d_out, 0, (size_t)N_GRAPHS * HID * sizeof(float), stream);
    pool_kernel<<<poolBlk, 256, 0, stream>>>(agg, batch, (float*)d_out, N_NODES);
}
